// KNNFeats_89928025243742
// MI455X (gfx1250) — matrix operand order repaired and checked by reading, not compiled, not run on hardware
//
#include <hip/hip_runtime.h>
#include <hip/hip_bf16.h>
#include <math.h>

// ---------------- types ----------------
typedef __attribute__((ext_vector_type(2)))  float  v2f;
typedef __attribute__((ext_vector_type(8)))  float  v8f;
typedef __attribute__((ext_vector_type(8)))  __bf16 v8bf;
typedef __attribute__((ext_vector_type(16))) __bf16 v16bf;
typedef __attribute__((ext_vector_type(4)))  unsigned int u32x4;
typedef __attribute__((ext_vector_type(8)))  int    i32x8;
typedef __attribute__((ext_vector_type(4)))  int    i32x4;

// ---------------- problem constants ----------------
constexpr int BB   = 8;
constexpr int CC   = 128;
constexpr int NN   = 2048;
constexpr int KK   = 8;
constexpr int C2   = 256;
constexpr int POS  = BB * NN * KK;     // 131072 positions

__device__ __forceinline__ __bf16 f2bf(float f) {
    unsigned int u = __builtin_bit_cast(unsigned int, f);
    unsigned int r = u + 0x7fffu + ((u >> 16) & 1u);   // round-to-nearest-even
    unsigned short h = (unsigned short)(r >> 16);
    return __builtin_bit_cast(__bf16, h);
}
__device__ __forceinline__ float bf2f(__bf16 b) {
    unsigned int u = ((unsigned int)__builtin_bit_cast(unsigned short, b)) << 16;
    return __builtin_bit_cast(float, u);
}

// ---------------- 1) xx = sum_c feats^2 ----------------
__global__ __launch_bounds__(256) void xx_kernel(const float* __restrict__ feats,
                                                 float* __restrict__ xx) {
    int t = blockIdx.x * 256 + threadIdx.x;       // (b, n)
    int b = t / NN, n = t % NN;
    const float* fb = feats + (size_t)b * CC * NN;
    float s = 0.f;
    for (int c = 0; c < CC; ++c) {
        float v = fb[(size_t)c * NN + n];
        s += v * v;
    }
    xx[t] = s;
}

// ---------------- 2) neg_dist via f32 WMMA 16x16x4 ----------------
__global__ __launch_bounds__(256) void dist_kernel(const float* __restrict__ feats,
                                                   const float* __restrict__ xx,
                                                   float* __restrict__ nd) {
    int wave = threadIdx.x >> 5, lane = threadIdx.x & 31;
    int tile = blockIdx.x * 8 + wave;             // 128*128 tiles per batch
    int b    = tile >> 14;
    int rem  = tile & 16383;
    int mBase = (rem >> 7) << 4;
    int nBase = (rem & 127) << 4;
    int half = lane >> 4, l16 = lane & 15;
    const float* fb = feats + (size_t)b * CC * NN;

    v8f acc = {};
    for (int kb = 0; kb < CC; kb += 4) {
        int k0 = kb + 2 * half;                   // A 16x4 f32: lanes<16 K=0,1 ; lanes>=16 K=2,3
        v2f a, bb;
        a[0]  = fb[(size_t)k0 * NN + mBase + l16];
        a[1]  = fb[(size_t)(k0 + 1) * NN + mBase + l16];
        bb[0] = fb[(size_t)k0 * NN + nBase + l16];
        bb[1] = fb[(size_t)(k0 + 1) * NN + nBase + l16];
        acc = __builtin_amdgcn_wmma_f32_16x16x4_f32(false, a, false, bb,
                                                    (short)0, acc, false, false);
    }
    const float* xb = xx + (size_t)b * NN;
    float xn = xb[nBase + l16];
    float* out = nd + (size_t)b * NN * NN;
#pragma unroll
    for (int r = 0; r < 8; ++r) {
        int m = mBase + r + 8 * half;             // D: vgpr r -> M=r (lanes<16) / r+8
        out[(size_t)m * NN + nBase + l16] = 2.0f * acc[r] - xb[m] - xn;
    }
}

// ---------------- 3) top-8 per row (wave per row) ----------------
__global__ __launch_bounds__(256) void topk_kernel(const float* __restrict__ nd,
                                                   int* __restrict__ idxOut) {
    __shared__ float sv[8][256];
    __shared__ int   si[8][256];
    int wave = threadIdx.x >> 5, lane = threadIdx.x & 31;
    int row = blockIdx.x * 8 + wave;              // b*N + m
    const float* r = nd + (size_t)row * NN;       // rows are contiguous across batches

    float vals[8]; int ids[8];
#pragma unroll
    for (int i = 0; i < 8; ++i) { vals[i] = -INFINITY; ids[i] = 0; }
    for (int c = lane; c < NN; c += 32) {
        float v = r[c];
        int   mn = 0; float mv = vals[0];
#pragma unroll
        for (int i = 1; i < 8; ++i) if (vals[i] < mv) { mv = vals[i]; mn = i; }
        if (v > mv) { vals[mn] = v; ids[mn] = c; }
    }
#pragma unroll
    for (int i = 0; i < 8; ++i) { sv[wave][lane * 8 + i] = vals[i]; si[wave][lane * 8 + i] = ids[i]; }
    __syncthreads();
    if (lane == 0) {
#pragma unroll 1
        for (int j = 0; j < 8; ++j) {
            float bv = -INFINITY; int bi = 1 << 30; int bp = 0;
#pragma unroll 1
            for (int t = 0; t < 256; ++t) {
                float v = sv[wave][t]; int id = si[wave][t];
                if (v > bv || (v == bv && id < bi)) { bv = v; bi = id; bp = t; }
            }
            sv[wave][bp] = -INFINITY;
            idxOut[(size_t)row * 8 + j] = bi;
        }
    }
}

// ---------------- 4) weights f32 -> bf16 ----------------
__global__ __launch_bounds__(256) void wcvt_kernel(const float* __restrict__ w1,
                                                   const float* __restrict__ w2,
                                                   const float* __restrict__ m1,
                                                   const float* __restrict__ m2,
                                                   __bf16* __restrict__ out) {
    int t = blockIdx.x * 256 + threadIdx.x;       // 0..131071
    float v;
    if      (t < 65536)  v = w1[t];
    else if (t < 98304)  v = w2[t - 65536];
    else if (t < 114688) v = m1[t - 98304];
    else                 v = m2[t - 114688];
    out[t] = f2bf(v);
}

// ---------------- 5) gather + attention softmax -> f (POS x 2C, bf16) --------
__global__ __launch_bounds__(256) void buildf_kernel(const float* __restrict__ feats,
                                                     const int*   __restrict__ idx,
                                                     const float* __restrict__ attn_w,
                                                     const float* __restrict__ attn_b,
                                                     __bf16* __restrict__ f) {
    int wave = threadIdx.x >> 5, lane = threadIdx.x & 31;
    int bn = blockIdx.x * 8 + wave;               // b*N + n
    int b = bn >> 11, n = bn & 2047;
    const float* fb = feats + (size_t)b * CC * NN;

    float ctr[4], aw[4], nbr[8][4];
#pragma unroll
    for (int j = 0; j < 4; ++j) {
        int c = lane + 32 * j;
        ctr[j] = fb[(size_t)c * NN + n];
        aw[j]  = attn_w[c];
    }
    int nid[8];
#pragma unroll
    for (int k = 0; k < 8; ++k) nid[k] = idx[(size_t)bn * 8 + k];
#pragma unroll
    for (int k = 0; k < 8; ++k)
#pragma unroll
        for (int j = 0; j < 4; ++j) {
            int c = lane + 32 * j;
            nbr[k][j] = fb[(size_t)c * NN + nid[k]];
        }
    float a[8];
#pragma unroll
    for (int k = 0; k < 8; ++k) {
        float p = 0.f;
#pragma unroll
        for (int j = 0; j < 4; ++j) p += aw[j] * (ctr[j] - nbr[k][j]);
#pragma unroll
        for (int m = 16; m >= 1; m >>= 1) p += __shfl_xor(p, m, 32);
        a[k] = p + attn_b[0];
    }
    float amax = a[0];
#pragma unroll
    for (int k = 1; k < 8; ++k) amax = fmaxf(amax, a[k]);
    float e[8], s = 0.f;
#pragma unroll
    for (int k = 0; k < 8; ++k) { e[k] = __expf(a[k] - amax); s += e[k]; }
    float inv = 1.0f / s;
#pragma unroll
    for (int k = 0; k < 8; ++k) {
        size_t base = ((size_t)bn * 8 + k) * C2;
        float w = e[k] * inv;
#pragma unroll
        for (int j = 0; j < 4; ++j) {
            int c = lane + 32 * j;
            f[base + c]       = f2bf(ctr[j]);
            f[base + 128 + c] = f2bf(w * nbr[k][j]);
        }
    }
}

// ---------------- 6) bf16 WMMA GEMM: Y(P,Dout) = X(P,Din) * W(Dout,Din)^T + bias
// Block: one 16-channel weight tile (TDM-staged into LDS), 8 waves x 64 positions.
template <int DIN, int DOUT>
__global__ __launch_bounds__(256) void gemm_kernel(const __bf16* __restrict__ X,
                                                   const __bf16* __restrict__ W,
                                                   const float*  __restrict__ bias,
                                                   __bf16* __restrict__ Y) {
    __shared__ __bf16 smemW[16 * DIN];            // 16 output channels x DIN (4-8 KB)
    constexpr int DT = DOUT / 16;
    int dTile  = blockIdx.x % DT;
    int pSuper = blockIdx.x / DT;
    int dBase  = dTile * 16;

    // ---- stage weight tile into LDS with the Tensor Data Mover ----
    if (threadIdx.x < 32) {                       // wave 0 issues (TDM ignores EXEC)
        unsigned long long ga = (unsigned long long)(uintptr_t)(W + (size_t)dBase * DIN);
        unsigned int lds = (unsigned int)(uintptr_t)smemW;   // low 32 bits = LDS byte offset
        u32x4 g0; i32x8 g1; i32x4 g2 = {}; i32x4 g3 = {}; i32x8 g4 = {};
        g0[0] = 1u;                                           // count=1, user mode
        g0[1] = lds;                                          // lds_addr
        g0[2] = (unsigned int)ga;                             // global_addr[31:0]
        g0[3] = (unsigned int)((ga >> 32) & 0x01FFFFFFull)    // global_addr[56:32]
              | (2u << 30);                                   // type=2 (image)
        g1[0] = 0x00010000;                                   // wg_mask=0, data_size=2B
        g1[1] = (int)(((unsigned)DIN & 0xffffu) << 16);       // tensor_dim0[15:0]
        g1[2] = (int)(16u << 16);                             // tensor_dim0 hi=0 | tensor_dim1=16
        g1[3] = (int)(((unsigned)DIN & 0xffffu) << 16);       // tensor_dim1 hi=0 | tile_dim0=DIN
        g1[4] = 16;                                           // tile_dim1=16, tile_dim2=0
        g1[5] = DIN;                                          // tensor_dim0_stride[31:0]
        g1[6] = 0; g1[7] = 0;
        __builtin_amdgcn_tensor_load_to_lds(g0, g1, g2, g3, g4, 0);
        __builtin_amdgcn_s_wait_tensorcnt(0);
    }
    __syncthreads();

    int wave = threadIdx.x >> 5, lane = threadIdx.x & 31;
    int half = lane >> 4, l16 = lane & 15;
    int pBase = (pSuper * 8 + wave) * 64;         // this wave: 64 consecutive positions

    const __bf16* wrow = smemW + (size_t)l16 * DIN + half * 16;
    const __bf16* x0   = X + (size_t)(pBase + l16) * DIN + half * 16;

    v8f acc[4] = {};
    for (int kb = 0; kb < DIN; kb += 32) {
        v8bf alo = *(const v8bf*)(wrow + kb);        // A: K = off..off+7     (off = half*16)
        v8bf ahi = *(const v8bf*)(wrow + kb + 8);    // A: K = off+8..off+15
        v16bf a;
#pragma unroll
        for (int i = 0; i < 8; ++i) { a[i] = alo[i]; a[8 + i] = ahi[i]; }
#pragma unroll
        for (int s = 0; s < 4; ++s) {                // 4 position-subtiles reuse A
            v16bf bfr = *(const v16bf*)(x0 + (size_t)s * 16 * DIN + kb);
            acc[s] = __builtin_amdgcn_wmma_f32_16x16x32_bf16(false, a, false, bfr,
                                                             (short)0, acc[s], false, false);
        }
    }
    int chBase = dBase + half * 8;
    float bvals[8];
#pragma unroll
    for (int r = 0; r < 8; ++r) bvals[r] = bias[chBase + r];
#pragma unroll
    for (int s = 0; s < 4; ++s) {
        int p = pBase + s * 16 + l16;
        v8bf outv;
#pragma unroll
        for (int r = 0; r < 8; ++r) outv[r] = f2bf(acc[s][r] + bvals[r]);
        *(v8bf*)(Y + (size_t)p * DOUT + chBase) = outv;
    }
}

// ---------------- 7) BN stats / apply ----------------
__global__ void zero_kernel(float* p, int n) {
    int t = blockIdx.x * blockDim.x + threadIdx.x;
    if (t < n) p[t] = 0.f;
}

template <int CH>
__global__ __launch_bounds__(256) void bnstats_kernel(const __bf16* __restrict__ X,
                                                      float* __restrict__ sums,
                                                      int posPerBlock) {
    constexpr int ROWS = 256 / CH;
    int ch = threadIdx.x % CH;
    int ro = threadIdx.x / CH;
    size_t p0 = (size_t)blockIdx.x * posPerBlock;
    float s1 = 0.f, s2 = 0.f;
    for (int i = ro; i < posPerBlock; i += ROWS) {
        float v = bf2f(X[(p0 + i) * CH + ch]);
        s1 += v; s2 += v * v;
    }
    atomicAdd(&sums[ch], s1);
    atomicAdd(&sums[CH + ch], s2);
}

template <int CH>
__global__ __launch_bounds__(256) void bnapply_kernel(const __bf16* __restrict__ X,
                                                      const float* __restrict__ sums,
                                                      const float* __restrict__ g,
                                                      const float* __restrict__ bb,
                                                      __bf16* __restrict__ Y) {
    size_t t = (size_t)blockIdx.x * 256 + threadIdx.x;
    int ch = (int)(t % CH);
    float mean = sums[ch] / (float)POS;
    float var  = sums[CH + ch] / (float)POS - mean * mean;
    float v = bf2f(X[t]);
    v = g[ch] * (v - mean) * rsqrtf(var + 1e-5f) + bb[ch];
    Y[t] = f2bf(fmaxf(v, 0.f));
}

// ---------------- 8) max over K -> d_out (B,C,N,1) f32 ----------------
__global__ __launch_bounds__(256) void maxk_kernel(const __bf16* __restrict__ m2,
                                                   float* __restrict__ out) {
    size_t t = (size_t)blockIdx.x * 256 + threadIdx.x;   // (b*N+n)*C + d
    int d = (int)(t % CC);
    size_t bn = t / CC;
    int b = (int)(bn >> 11), n = (int)(bn & 2047);
    float mv = -INFINITY;
#pragma unroll
    for (int k = 0; k < 8; ++k)
        mv = fmaxf(mv, bf2f(m2[(bn * 8 + k) * CC + d]));
    out[((size_t)b * CC + d) * NN + n] = mv;
}

// ---------------- launch ----------------
extern "C" void kernel_launch(void* const* d_in, const int* in_sizes, int n_in,
                              void* d_out, int out_size, void* d_ws, size_t ws_size,
                              hipStream_t stream) {
    const float* feats  = (const float*)d_in[0];
    const float* attn_w = (const float*)d_in[1];
    const float* attn_b = (const float*)d_in[2];
    const float* cf_w1  = (const float*)d_in[3];
    const float* cf_b1  = (const float*)d_in[4];
    const float* bn1_g  = (const float*)d_in[5];
    const float* bn1_b  = (const float*)d_in[6];
    const float* cf_w2  = (const float*)d_in[7];
    const float* cf_b2  = (const float*)d_in[8];
    const float* mlp_w1 = (const float*)d_in[9];
    const float* mlp_b1 = (const float*)d_in[10];
    const float* bn2_g  = (const float*)d_in[11];
    const float* bn2_b  = (const float*)d_in[12];
    const float* mlp_w2 = (const float*)d_in[13];
    const float* mlp_b2 = (const float*)d_in[14];
    float* out = (float*)d_out;

    char* ws = (char*)d_ws;
    // region A: 134 MB (neg_dist; later h1pre@0 / h1n@+67MB; later m1pre@0 / m1n@+67MB)
    const size_t offA     = 0;
    const size_t offSmall = 134217728;
    const size_t off_xx   = offSmall;                    // 65536 B
    const size_t off_idx  = offSmall + 65536;            // 524288 B
    const size_t off_w    = offSmall + 65536 + 524288;   // 262144 B (bf16 weights)
    const size_t off_st   = off_w + 262144;              // 2048 B (BN sums)
    const size_t offB     = offSmall + 1048576;          // 67 MB region (f; later h2@0, m2@+33.5MB)

    float*  nd    = (float*)(ws + offA);
    float*  xx    = (float*)(ws + off_xx);
    int*    idx   = (int*)  (ws + off_idx);
    __bf16* wbf   = (__bf16*)(ws + off_w);
    float*  stats = (float*)(ws + off_st);
    __bf16* fmat  = (__bf16*)(ws + offB);
    __bf16* h1pre = (__bf16*)(ws + offA);
    __bf16* h1n   = (__bf16*)(ws + offA + 67108864);
    __bf16* h2    = (__bf16*)(ws + offB);
    __bf16* m1pre = (__bf16*)(ws + offA);
    __bf16* m1n   = (__bf16*)(ws + offA + 67108864);
    __bf16* m2    = (__bf16*)(ws + offB + 33554432);

    __bf16* wbf1  = wbf;             // 256*256
    __bf16* wbf2  = wbf + 65536;     // 128*256
    __bf16* wbfm1 = wbf + 98304;     // 128*128
    __bf16* wbfm2 = wbf + 114688;    // 128*128

    xx_kernel   <<<64,    256, 0, stream>>>(feats, xx);
    dist_kernel <<<16384, 256, 0, stream>>>(feats, xx, nd);
    topk_kernel <<<2048,  256, 0, stream>>>(nd, idx);
    wcvt_kernel <<<512,   256, 0, stream>>>(cf_w1, cf_w2, mlp_w1, mlp_w2, wbf);
    buildf_kernel<<<2048, 256, 0, stream>>>(feats, idx, attn_w, attn_b, fmat);

    // POS/512 = 256 position-supergroups per d-tile
    gemm_kernel<256,256><<<256 * 16, 256, 0, stream>>>(fmat, wbf1, cf_b1, h1pre);
    zero_kernel<<<2, 256, 0, stream>>>(stats, 512);
    bnstats_kernel<256><<<512, 256, 0, stream>>>(h1pre, stats, POS / 512);
    bnapply_kernel<256><<<(POS * 256) / 256, 256, 0, stream>>>(h1pre, stats, bn1_g, bn1_b, h1n);

    gemm_kernel<256,128><<<256 * 8, 256, 0, stream>>>(h1n, wbf2, cf_b2, h2);
    gemm_kernel<128,128><<<256 * 8, 256, 0, stream>>>(h2, wbfm1, mlp_b1, m1pre);

    zero_kernel<<<2, 256, 0, stream>>>(stats, 512);
    bnstats_kernel<128><<<512, 256, 0, stream>>>(m1pre, stats, POS / 512);
    bnapply_kernel<128><<<(POS * 128) / 256, 256, 0, stream>>>(m1pre, stats, bn2_g, bn2_b, m1n);

    gemm_kernel<128,128><<<256 * 8, 256, 0, stream>>>(m1n, wbfm2, mlp_b2, m2);
    maxk_kernel<<<(BB * CC * NN) / 256, 256, 0, stream>>>(m2, out);
}
